// ContrastiveModel_16612933501072
// MI455X (gfx1250) — matrix operand order repaired and checked by reading, not compiled, not run on hardware
//
#include <hip/hip_runtime.h>

typedef __attribute__((ext_vector_type(16))) _Float16 v16h;
typedef __attribute__((ext_vector_type(8)))  _Float16 v8h;
typedef __attribute__((ext_vector_type(4)))  _Float16 v4h;
typedef __attribute__((ext_vector_type(8)))  float    v8f;

#define TEMP_INV (1.0f / 0.07f)

// ---- WMMA fragment loaders (layouts per CDNA5 ISA 7.12.2) ----

// A-matrix 16x32 f16, built from an f32 K-contiguous row.
// lanes 0-15:  e0..7 = K0..7,  e8..15 = K16..23
// lanes 16-31: e0..7 = K8..15, e8..15 = K24..31
__device__ __forceinline__ v16h load_a_f32(const float* __restrict__ row, int k0, int half) {
  const float* p0 = row + k0 + 8 * half;
  const float* p1 = row + k0 + 16 + 8 * half;
  float4 x0 = *(const float4*)(p0);
  float4 x1 = *(const float4*)(p0 + 4);
  float4 y0 = *(const float4*)(p1);
  float4 y1 = *(const float4*)(p1 + 4);
  v16h r;
  r[0]  = (_Float16)x0.x; r[1]  = (_Float16)x0.y; r[2]  = (_Float16)x0.z; r[3]  = (_Float16)x0.w;
  r[4]  = (_Float16)x1.x; r[5]  = (_Float16)x1.y; r[6]  = (_Float16)x1.z; r[7]  = (_Float16)x1.w;
  r[8]  = (_Float16)y0.x; r[9]  = (_Float16)y0.y; r[10] = (_Float16)y0.z; r[11] = (_Float16)y0.w;
  r[12] = (_Float16)y1.x; r[13] = (_Float16)y1.y; r[14] = (_Float16)y1.z; r[15] = (_Float16)y1.w;
  return r;
}

// B-matrix 32x16 f16 from an f16 K-contiguous row (one column of B per lane).
// lanes 0-15: e0..7 = K0..7, e8..15 = K16..23 ; lanes 16-31: e0..7 = K8..15, e8..15 = K24..31
__device__ __forceinline__ v16h load_b_f16(const _Float16* __restrict__ row, int k0, int half) {
  v8h a = *(const v8h*)(row + k0 + 8 * half);
  v8h b = *(const v8h*)(row + k0 + 16 + 8 * half);
  v16h r;
#pragma unroll
  for (int i = 0; i < 8; ++i) { r[i] = a[i]; r[8 + i] = b[i]; }
  return r;
}

// ---- Kernel 0: convert W (512x512 f32 -> f16), L2-resident operand for kernel 1 ----
__global__ __launch_bounds__(256) void cvt_w_kernel(const float* __restrict__ W,
                                                    _Float16* __restrict__ W16) {
  int idx = (blockIdx.x * 256 + threadIdx.x) * 4;   // grid covers 512*512
  float4 v = *(const float4*)(W + idx);
  v4h o;
  o[0] = (_Float16)v.x; o[1] = (_Float16)v.y; o[2] = (_Float16)v.z; o[3] = (_Float16)v.w;
  *(v4h*)(W16 + idx) = o;
}

// ---- Kernel 1: pred[m,l,c] = sum_p P[m,l,p]*W[c,p] + b[c], stored f16 ----
// grid (64 l-tiles, 128 m), block 128 (4 waves). Wave w: c-range [128w,128w+128).
__global__ __launch_bounds__(128) void pred_gemm_kernel(
    const float* __restrict__ P,        // (128,1024,512) f32
    const _Float16* __restrict__ W16,   // (512,512) f16, row-major [c][p]
    const float* __restrict__ bias,     // (512) f32
    _Float16* __restrict__ pred,        // (128, Lp, 512) f16
    int Lp) {
  const int tid = threadIdx.x;
  const int w = tid >> 5;
  const int lane = tid & 31;
  const int laneLow = lane & 15;
  const int half = lane >> 4;
  const int n = blockIdx.y;
  const int l0 = blockIdx.x * 16;
  const int cbase = w * 128;

  const float* Arow = P + ((size_t)n * 1024 + (size_t)(l0 + laneLow)) * 512;
  const _Float16* Wrow = W16 + (size_t)(cbase + laneLow) * 512;

  v8f acc[8] = {};
  for (int k0 = 0; k0 < 512; k0 += 32) {
    // Stage: issue all fragment loads first so WMMAs overlap the load tail.
    v16h a = load_a_f32(Arow, k0, half);
    v16h b[8];
#pragma unroll
    for (int t = 0; t < 8; ++t)
      b[t] = load_b_f16(Wrow + (size_t)t * 16 * 512, k0, half);
#pragma unroll
    for (int t = 0; t < 8; ++t)
      acc[t] = __builtin_amdgcn_wmma_f32_16x16x32_f16(false, a, false, b[t],
                                                      (short)0, acc[t], false, false);
  }

#pragma unroll
  for (int t = 0; t < 8; ++t) {
    int c = cbase + t * 16 + laneLow;
    float bv = bias[c];
#pragma unroll
    for (int r = 0; r < 8; ++r) {
      int l = l0 + r + 8 * half;               // D: row m = r + 8*half
      if (l < Lp)
        pred[((size_t)n * Lp + l) * 512 + c] = (_Float16)(acc[t][r] + bv);
    }
  }
}

// ---- Kernel 2: l_neg[n,m,l] = sum_c T[n,l+step,c]*pred[m,l,c]; logits + NCE acc ----
// grid (128 l-blocks of 8, 8 n-tiles), block 256 (8 waves; wave w = m-tile).
__global__ __launch_bounds__(256) void logits_kernel(
    const float* __restrict__ T,        // (128,1024,512) f32
    const _Float16* __restrict__ pred,  // (128, Lp, 512) f16
    float* __restrict__ out,            // logits base (128,128,2045)
    unsigned* __restrict__ gcount,
    int Lp, int step, int off) {
  __shared__ float stage[1024 * 9];     // 16 n x 64 m x 8 l, stride-9 pad
  __shared__ float posLDS[8][16];
  __shared__ unsigned badLDS[128];
  __shared__ unsigned cntLDS;

  const int tid = threadIdx.x;
  const int w = tid >> 5;
  const int lane = tid & 31;
  const int laneLow = lane & 15;
  const int half = lane >> 4;
  const int bn = blockIdx.y;
  const int l0 = blockIdx.x * 8;

  const float* Ap[8];
  const _Float16* Bp[8];
#pragma unroll
  for (int li = 0; li < 8; ++li) {
    int lA = l0 + li + step; if (lA > 1023) lA = 1023;       // clamp ragged edge
    int lB = l0 + li;        if (lB > Lp - 1) lB = Lp - 1;
    Ap[li] = T + ((size_t)(bn * 16 + laneLow) * 1024 + lA) * 512;
    Bp[li] = pred + ((size_t)(w * 16 + laneLow) * Lp + lB) * 512;
  }

  v8f acc[8] = {};
  for (int k0 = 0; k0 < 512; k0 += 32) {
    // Stage fragments in groups of 4 li: 24 independent loads ahead of each
    // WMMA group, without blowing the VGPR budget.
#pragma unroll
    for (int grp = 0; grp < 2; ++grp) {
      v16h a[4], b[4];
#pragma unroll
      for (int u = 0; u < 4; ++u) {
        int li = grp * 4 + u;
        a[u] = load_a_f32(Ap[li], k0, half);
        b[u] = load_b_f16(Bp[li], k0, half);
      }
#pragma unroll
      for (int u = 0; u < 4; ++u) {
        int li = grp * 4 + u;
        acc[li] = __builtin_amdgcn_wmma_f32_16x16x32_f16(false, a[u], false, b[u],
                                                         (short)0, acc[li], false, false);
      }
    }
  }

  // --- NCE accuracy bookkeeping ---
  if (tid < 128) badLDS[tid] = 0;
  if (tid == 0) cntLDS = 0;
  if (w == bn) {                         // this wave's m-tile holds the diagonal
#pragma unroll
    for (int li = 0; li < 8; ++li)
#pragma unroll
      for (int r = 0; r < 8; ++r)
        if (laneLow == r + 8 * half)     // element where m_local == n_local
          posLDS[li][laneLow] = acc[li][r];
  }
  __syncthreads();

#pragma unroll
  for (int li = 0; li < 8; ++li) {
#pragma unroll
    for (int r = 0; r < 8; ++r) {
      int nl = r + 8 * half;
      float pos = posLDS[li][nl];
      int mg = w * 16 + laneLow;
      int ng = bn * 16 + nl;
      if (mg != ng && acc[li][r] >= pos) badLDS[li * 16 + nl] = 1u;
    }
  }
  __syncthreads();

  if (tid < 128) {
    int li = tid >> 4;
    if ((l0 + li) < Lp && badLDS[tid] == 0) atomicAdd(&cntLDS, 1u);
  }

  // --- logits store: transpose through LDS so l runs are contiguous ---
#pragma unroll
  for (int ph = 0; ph < 2; ++ph) {
    __syncthreads();
    if ((w >> 2) == ph) {
      int mh = (w & 3) * 16 + laneLow;
#pragma unroll
      for (int li = 0; li < 8; ++li)
#pragma unroll
        for (int r = 0; r < 8; ++r) {
          int nl = r + 8 * half;
          stage[(nl * 64 + mh) * 9 + li] = acc[li][r] * TEMP_INV;
        }
    }
    __syncthreads();
    int g = tid >> 3;          // 32 groups of 8 threads
    int li2 = tid & 7;
    int l = l0 + li2;
    if (l < Lp) {
      for (int j = 0; j < 32; ++j) {
        int pi = g + 32 * j;   // 1024 (n,m) pairs, conflict-free banks
        int nl = pi >> 6, mh = pi & 63;
        int ng = bn * 16 + nl;
        int mg = ph * 64 + mh;
        out[((size_t)(ng * 128 + mg)) * 2045 + off + l] = stage[pi * 9 + li2];
      }
    }
  }
  __syncthreads();
  if (tid == 0 && cntLDS) atomicAdd(gcount, cntLDS);
}

// ---- Kernel 3: y_trues + accs ----
__global__ void finalize_kernel(float* __restrict__ out, const unsigned* __restrict__ cnt) {
  const size_t LOGITS = (size_t)128 * 128 * 2045;
  int idx = blockIdx.x * 256 + threadIdx.x;
  if (idx < 128 * 2045)
    out[LOGITS + idx] = (float)(idx / 2045);
  if (idx == 0) {
    out[LOGITS + (size_t)128 * 2045 + 0] = (float)cnt[0] / (128.0f * 1023.0f);
    out[LOGITS + (size_t)128 * 2045 + 1] = (float)cnt[1] / (128.0f * 1022.0f);
  }
}

extern "C" void kernel_launch(void* const* d_in, const int* in_sizes, int n_in,
                              void* d_out, int out_size, void* d_ws, size_t ws_size,
                              hipStream_t stream) {
  const float* T  = (const float*)d_in[0];
  const float* P  = (const float*)d_in[1];
  const float* W1 = (const float*)d_in[2];
  const float* b1 = (const float*)d_in[3];
  const float* W2 = (const float*)d_in[4];
  const float* b2 = (const float*)d_in[5];
  float* out = (float*)d_out;

  // ws layout: [0,256) counters | pred f16 (128*1023*512) | W16 f16 (512*512)
  unsigned* cnt = (unsigned*)d_ws;
  _Float16* pred = (_Float16*)((char*)d_ws + 256);
  _Float16* W16  = (_Float16*)((char*)d_ws + 256 + (size_t)128 * 1023 * 512 * 2);

  hipMemsetAsync(d_ws, 0, 256, stream);

  // step 1 (Lp = 1023)
  cvt_w_kernel<<<dim3(256), 256, 0, stream>>>(W1, W16);
  pred_gemm_kernel<<<dim3(64, 128), 128, 0, stream>>>(P, W16, b1, pred, 1023);
  logits_kernel<<<dim3(128, 8), 256, 0, stream>>>(T, pred, out, cnt + 0, 1023, 1, 0);
  // step 2 (Lp = 1022), pred + W16 buffers reused (stream-ordered)
  cvt_w_kernel<<<dim3(256), 256, 0, stream>>>(W2, W16);
  pred_gemm_kernel<<<dim3(64, 128), 128, 0, stream>>>(P, W16, b2, pred, 1022);
  logits_kernel<<<dim3(128, 8), 256, 0, stream>>>(T, pred, out, cnt + 1, 1022, 2, 1023);

  finalize_kernel<<<dim3((128 * 2045 + 255) / 256), 256, 0, stream>>>(out, cnt);
}
